// TransformerModuleNet_14894946582997
// MI455X (gfx1250) — compile-verified, matrix operand order checked by reading, not run on hardware
//
#include <hip/hip_runtime.h>
#include <hip/hip_bf16.h>
#include <math.h>

// ---------------- problem constants ----------------
#define BSZ    8
#define RR     64
#define VFD    2048
#define HID    768
#define NHEAD  12
#define DHEAD  64
#define FFD    3072
#define NMOD   12
#define NLAY   2
#define NSTEP  12
#define NPROG  26
#define NARG   46
#define NLABEL 32
#define SREAL  66     // R + 2
#define SPADR  80     // padded sequence rows (5 x 16)
#define SLD    96     // scores leading dim (3 x 32: ctx K-loop needs no guards)

__constant__ int c_func_map[NPROG] =
    {0,1,2,3,4,5,6,7,7,8,7,8,8,8,9,9,9,9,10,10,10,10,11,11,11,11};

typedef __bf16 bf16;
typedef __bf16 v16bf __attribute__((ext_vector_type(16)));
typedef __bf16 v2bf  __attribute__((ext_vector_type(2)));
typedef float  v8f   __attribute__((ext_vector_type(8)));
typedef unsigned int u32;

#if defined(__AMDGCN__) && __has_builtin(__builtin_amdgcn_tensor_load_to_lds)
#define HAVE_TDM 1
#else
#define HAVE_TDM 0
#endif

// pack two floats to packed-bf16 dword (low = first) -> v_cvt_pk_bf16_f32
__device__ __forceinline__ u32 pkbf(float a, float b) {
    v2bf t;
    t[0] = (bf16)a;
    t[1] = (bf16)b;
    return __builtin_bit_cast(u32, t);
}

#if HAVE_TDM
typedef unsigned int u32x4 __attribute__((ext_vector_type(4)));
typedef int          i32x4 __attribute__((ext_vector_type(4)));
typedef int          i32x8 __attribute__((ext_vector_type(8)));

__device__ __forceinline__ void tdm_wait0() {
#if __has_builtin(__builtin_amdgcn_s_wait_tensorcnt)
    __builtin_amdgcn_s_wait_tensorcnt((short)0);
#else
    asm volatile("s_wait_tensorcnt 0" ::: "memory");
#endif
}

// TDM: DMA a (16 rows x 128 dwords) tile of pre-packed bf16 weights into LDS,
// padding 4 dwords after every 128 (-> 132-dword LDS row stride).
// D# fields per CDNA5 ISA.
__device__ __forceinline__ void tdm_load_b_tile(const u32* __restrict__ src, int N,
                                                void* dstLds) {
    unsigned long long ga = (unsigned long long)(uintptr_t)src;
    u32 lds = (u32)(uintptr_t)dstLds;
    u32x4 g0;
    g0[0] = 1u;                                            // count=1 (valid descriptor)
    g0[1] = lds;                                           // lds_addr
    g0[2] = (u32)ga;                                       // global_addr[31:0]
    g0[3] = (u32)((ga >> 32) & 0x1ffffffu) | (2u << 30);   // global_addr[56:32] | type=2
    i32x8 g1;
    g1[0] = (2 << 16) | (1 << 20) | (6 << 22) | (3 << 25); // 4B elems; pad 4dw per 128dw
    g1[1] = (int)((N & 0xffff) << 16);                     // tensor_dim0 lo16
    g1[2] = (int)(((N >> 16) & 0xffff) | (16 << 16));      // tensor_dim0 hi16 | dim1=16
    g1[3] = (int)(128 << 16);                              // tile_dim0 = 128 dwords
    g1[4] = 16;                                            // tile_dim1 = 16 rows
    g1[5] = N;                                             // tensor_dim0_stride lo32
    g1[6] = 0;
    g1[7] = 0;
    i32x4 z4 = {0, 0, 0, 0};
#if __clang_major__ >= 23
    i32x8 z8 = {0, 0, 0, 0, 0, 0, 0, 0};
    __builtin_amdgcn_tensor_load_to_lds(g0, g1, z4, z4, z8, 0);
#else
    __builtin_amdgcn_tensor_load_to_lds(g0, g1, z4, z4, 0);
#endif
}
#endif  // HAVE_TDM

// ---------------- fragment loaders ----------------
// CDNA5 16-bit A layout: lanes 0-15 hold rows, VGPR v holds K pair
// (v<4: K=2v, v>=4: K=16+2(v-4)); lanes 16-31 at K+8 => pair offset p0 = 0|4.
__device__ __forceinline__ v16bf load_fa(const u32 (*As4)[28], int row, int p0) {
    union { uint4 q[2]; v16bf v; } fa;
    fa.q[0] = *(const uint4*)&As4[row][p0];
    fa.q[1] = *(const uint4*)&As4[row][p0 + 8];
    return fa.v;
}
template <int STRIDE>
__device__ __forceinline__ v16bf load_fb(const u32 (*Bs2)[STRIDE], int col, int p0) {
    union { u32 u[8]; v16bf v; } fb;
#pragma unroll
    for (int v = 0; v < 4; ++v) {
        fb.u[v]     = Bs2[p0 + v][col];
        fb.u[v + 4] = Bs2[p0 + 8 + v][col];
    }
    return fb.v;
}
__device__ __forceinline__ v8f wmma_bf16(v16bf fa, v16bf fb, v8f acc) {
    return __builtin_amdgcn_wmma_f32_16x16x32_bf16(false, fa, false, fb,
                                                   (short)0, acc, false, false);
}

// ---------------- weight pre-pack: fp32 (rows x N) -> K-pair-packed bf16 ----------
__global__ void __launch_bounds__(256)
pack_weights_kernel(const float* __restrict__ W, u32* __restrict__ Wp,
                    long rows, int N) {
    long total = (rows >> 1) * (long)(N >> 2);
    for (long i = (long)blockIdx.x * 256 + threadIdx.x; i < total;
         i += (long)gridDim.x * 256) {
        long p  = i / (N >> 2);
        int  c4 = (int)(i % (N >> 2)) << 2;
        const float* b0 = W + p * 2 * (long)N + c4;
        const float4 f0 = *(const float4*)b0;
        const float4 f1 = *(const float4*)(b0 + N);
        uint4 u;
        u.x = pkbf(f0.x, f1.x);
        u.y = pkbf(f0.y, f1.y);
        u.z = pkbf(f0.z, f1.z);
        u.w = pkbf(f0.w, f1.w);
        *(uint4*)(Wp + p * (long)N + c4) = u;
    }
}

// ---------------- main GEMM: 256 threads, 32(M) x 128(N) block tile ----------------
// 8 waves in 2(M) x 4(N); each wave computes two adjacent 16x16 tiles sharing one
// A fragment => 16 WMMA per staged (A 32x32 + B 32x128) K-step.
// B staging: TDM (tensor_load_to_lds) from pre-packed bf16 if available, else
// b128 copy of packed bf16, else fp32 convert fallback.
__global__ void __launch_bounds__(256)
wmma_gemm_kernel(const float* __restrict__ Abase, long aBS, int lda,
                 int Mreal, int Marows,
                 const float* __restrict__ Wbase, const u32* __restrict__ Wpbase,
                 long wSel, int N, int Kpad,
                 const float* __restrict__ biasBase, long bSel,
                 const float* __restrict__ bias2,
                 const float* __restrict__ residBase, long rBS,
                 const float* __restrict__ spat, const float* __restrict__ Wloc,
                 float* __restrict__ Ybase, long yBS, int ldy,
                 const int* __restrict__ args, int step, int lsel, int do_gelu) {
    __shared__ u32 As4[2][32][28];
    __shared__ u32 Bs2[2][16][132];

    const int b = blockIdx.z;
    const float* A     = Abase + (long)b * aBS;
    const float* resid = residBase ? residBase + (long)b * rBS : nullptr;
    float*       Y     = Ybase + (long)b * yBS;
    const float* W     = Wbase;
    const u32*   Wp    = Wpbase;
    const float* bias  = biasBase;
    if (args) {
        int prog = args[(b * NSTEP + step) * 2];
        prog = prog < 0 ? 0 : (prog >= NPROG ? NPROG - 1 : prog);
        long sel = (long)c_func_map[prog] * NLAY + lsel;
        W += sel * wSel;
        if (Wp) Wp += sel * (wSel >> 1);
        if (bias) bias += sel * bSel;
    }
    const int m0   = blockIdx.y * 32;
    const int n0   = blockIdx.x * 128;
    const int tid  = threadIdx.x;
    const int lane = tid & 31;
    const int w    = tid >> 5;
    const int wm   = w & 1;        // M half (0|1)
    const int wn   = w >> 1;       // N quarter, owns cols [wn*32, wn*32+32)
    const int nk   = Kpad >> 5;

    // stage A: 256 tasks = 32 rows x 8 float4 (1/thread); clamp addr, mask rows >= Mreal
    auto stageA = [&](u32 (*dst)[28], int k0) {
        int row = tid >> 3;
        int c4  = (tid & 7) * 4;
        int gr  = m0 + row;
        int grc = gr < Marows - 1 ? gr : Marows - 1;
        float m = (gr < Mreal) ? 1.0f : 0.0f;
        const float4 f = *(const float4*)(A + (long)grc * lda + (k0 + c4));
        *(uint2*)&dst[row][c4 >> 1] =
            make_uint2(pkbf(f.x * m, f.y * m), pkbf(f.z * m, f.w * m));
    };
    // issue B-tile staging for buffer `buf`, K offset k0
    auto stageB_issue = [&](int buf, int k0) {
#if HAVE_TDM
        if (Wp) {  // one TDM DMA per K-step, issued by wave 0 only (TENSORcnt-tracked)
            if (w == 0)
                tdm_load_b_tile(Wp + ((long)(k0 >> 1)) * N + n0, N, &Bs2[buf][0][0]);
            return;
        }
#endif
        if (Wp) {  // pre-packed bf16: straight b128 copy
#pragma unroll
            for (int j = 0; j < 2; ++j) {
                int t  = tid + j * 256;
                int rp = t >> 5;
                int c4 = (t & 31) * 4;
                const uint4 u = *(const uint4*)(Wp + ((long)(k0 >> 1) + rp) * N + (n0 + c4));
                *(uint4*)&Bs2[buf][rp][c4] = u;
            }
        } else {   // fp32 fallback: load two rows, convert+pack
#pragma unroll
            for (int j = 0; j < 2; ++j) {
                int t  = tid + j * 256;
                int rp = t >> 5;
                int c4 = (t & 31) * 4;
                const float* p = W + (long)(k0 + 2 * rp) * N + (n0 + c4);
                const float4 f0 = *(const float4*)p;
                const float4 f1 = *(const float4*)(p + N);
                uint4 u;
                u.x = pkbf(f0.x, f1.x);
                u.y = pkbf(f0.y, f1.y);
                u.z = pkbf(f0.z, f1.z);
                u.w = pkbf(f0.w, f1.w);
                *(uint4*)&Bs2[buf][rp][c4] = u;
            }
        }
    };
    auto stageB_wait = [&]() {   // drain outstanding TDM before the barrier
#if HAVE_TDM
        if (Wp && w == 0) tdm_wait0();
#endif
    };

    stageB_issue(0, 0);
    stageA(As4[0], 0);
    stageB_wait();
    __syncthreads();

    v8f acc0 = {}, acc1 = {};
    const int lr = lane & 15;
    const int p0 = (lane & 16) ? 4 : 0;
    int cur = 0;
    for (int kt = 0; kt < nk; ++kt) {
        if (kt + 1 < nk) {
            stageB_issue(cur ^ 1, (kt + 1) << 5);
            stageA(As4[cur ^ 1], (kt + 1) << 5);
        }
        v16bf fa  = load_fa(As4[cur], wm * 16 + lr, p0);
        v16bf fb0 = load_fb<132>(Bs2[cur], wn * 32 + lr, p0);
        v16bf fb1 = load_fb<132>(Bs2[cur], wn * 32 + 16 + lr, p0);
        acc0 = wmma_bf16(fa, fb0, acc0);
        acc1 = wmma_bf16(fa, fb1, acc1);
        if (kt + 1 < nk) stageB_wait();
        __syncthreads();
        cur ^= 1;
    }

    int colg0 = n0 + wn * 32 + lr;
    int colg1 = colg0 + 16;
    int rbase = (lane & 16) ? 8 : 0;
    float badd0 = bias ? bias[colg0] : 0.0f;
    float badd1 = bias ? bias[colg1] : 0.0f;
    if (bias2) { badd0 += bias2[colg0]; badd1 += bias2[colg1]; }
#pragma unroll
    for (int r = 0; r < 8; ++r) {
        int rowg = m0 + wm * 16 + rbase + r;
        if (rowg < Marows) {
            float v0 = acc0[r] + badd0;
            float v1 = acc1[r] + badd1;
            if (spat) {
                const float* sp = spat + (long)rowg * 5;
#pragma unroll
                for (int j = 0; j < 5; ++j) {
                    v0 += sp[j] * Wloc[j * HID + colg0];
                    v1 += sp[j] * Wloc[j * HID + colg1];
                }
            }
            if (resid) {
                v0 += resid[(long)rowg * ldy + colg0];
                v1 += resid[(long)rowg * ldy + colg1];
            }
            if (do_gelu) {
                v0 = 0.5f * v0 * (1.0f + erff(v0 * 0.70710678118654752f));
                v1 = 0.5f * v1 * (1.0f + erff(v1 * 0.70710678118654752f));
            }
            Y[(long)rowg * ldy + colg0] = v0;
            Y[(long)rowg * ldy + colg1] = v1;
        }
    }
}

// ---------------- attention scores ----------------
__global__ void __launch_bounds__(128)
attn_scores_kernel(const float* __restrict__ q, const float* __restrict__ k,
                   float* __restrict__ scores, float scale) {
    __shared__ u32 As4[2][16][28];
    __shared__ u32 Bs2[2][16][68];
    const int bh = blockIdx.z;
    const int b = bh / NHEAD, hh = bh % NHEAD;
    const float* Q = q + (long)b * SPADR * HID + hh * DHEAD;
    const float* K = k + (long)b * SPADR * HID + hh * DHEAD;
    float* S = scores + (long)bh * SPADR * SLD;
    const int m0 = blockIdx.y * 16, n0 = blockIdx.x * 64;
    const int tid = threadIdx.x, lane = tid & 31, w = tid >> 5;

    auto stageA = [&](u32 (*dst)[28], int k0) {
        int row = tid >> 3;
        int c4  = (tid & 7) * 4;
        int gr  = m0 + row;
        float m = (gr < SREAL) ? 1.0f : 0.0f;
        const float4 f = *(const float4*)(Q + (long)gr * HID + (k0 + c4));
        *(uint2*)&dst[row][c4 >> 1] =
            make_uint2(pkbf(f.x * m, f.y * m), pkbf(f.z * m, f.w * m));
    };
    auto stageB = [&](u32 (*dst)[68], int k0) {
#pragma unroll
        for (int j = 0; j < 4; ++j) {
            int t  = tid + j * 128;
            int qd = t & 7;
            int c  = t >> 3;
            int gn = n0 + c;
            int gnc = gn < SPADR - 1 ? gn : SPADR - 1;
            float m = (gn < SREAL) ? 1.0f : 0.0f;
            const float4 f = *(const float4*)(K + (long)gnc * HID + k0 + 4 * qd);
            dst[2 * qd][c]     = pkbf(f.x * m, f.y * m);
            dst[2 * qd + 1][c] = pkbf(f.z * m, f.w * m);
        }
    };

    stageA(As4[0], 0);
    stageB(Bs2[0], 0);
    __syncthreads();
    v8f acc = {};
    const int lr = lane & 15;
    const int p0 = (lane & 16) ? 4 : 0;
    int cur = 0;
    for (int kt = 0; kt < 2; ++kt) {
        if (kt == 0) { stageA(As4[1], 32); stageB(Bs2[1], 32); }
        v16bf fa = load_fa(As4[cur], lr, p0);
        v16bf fb = load_fb<68>(Bs2[cur], w * 16 + lr, p0);
        acc = wmma_bf16(fa, fb, acc);
        __syncthreads();
        cur ^= 1;
    }
    int colg  = n0 + w * 16 + lr;
    int rbase = (lane & 16) ? 8 : 0;
    if (colg < SLD) {
#pragma unroll
        for (int r = 0; r < 8; ++r) {
            int rowg = m0 + rbase + r;
            S[(long)rowg * SLD + colg] = acc[r] * scale;
        }
    }
}

__global__ void __launch_bounds__(128)
softmax_kernel(float* __restrict__ scores) {
    const int row = blockIdx.x;
    const int bh  = blockIdx.y;
    float* Srow = scores + (long)bh * SPADR * SLD + (long)row * SLD;
    __shared__ float red[128];
    const int tid = threadIdx.x;
    float v = (tid < SREAL) ? Srow[tid] : -1e30f;
    red[tid] = v;
    __syncthreads();
    for (int off = 64; off > 0; off >>= 1) {
        if (tid < off) red[tid] = fmaxf(red[tid], red[tid + off]);
        __syncthreads();
    }
    float mx = red[0];
    __syncthreads();
    float e = (tid < SREAL) ? __expf(v - mx) : 0.0f;
    red[tid] = e;
    __syncthreads();
    for (int off = 64; off > 0; off >>= 1) {
        if (tid < off) red[tid] += red[tid + off];
        __syncthreads();
    }
    float s = red[0];
    if (tid < SLD) Srow[tid] = (tid < SREAL) ? (e / s) : 0.0f;
}

__global__ void __launch_bounds__(128)
attn_ctx_kernel(const float* __restrict__ scores, const float* __restrict__ v,
                float* __restrict__ ctx) {
    __shared__ u32 As4[2][16][28];
    __shared__ u32 Bs2[2][16][68];
    const int bh = blockIdx.z;
    const int b = bh / NHEAD, hh = bh % NHEAD;
    const float* S = scores + (long)bh * SPADR * SLD;
    const float* V = v   + (long)b * SPADR * HID + hh * DHEAD;
    float*       C = ctx + (long)b * SPADR * HID + hh * DHEAD;
    const int m0 = blockIdx.y * 16;
    const int tid = threadIdx.x, lane = tid & 31, w = tid >> 5;

    auto stageA = [&](u32 (*dst)[28], int k0) {
        int row = tid >> 3;
        int c4  = (tid & 7) * 4;
        const float4 f = *(const float4*)(S + (long)(m0 + row) * SLD + (k0 + c4));
        *(uint2*)&dst[row][c4 >> 1] = make_uint2(pkbf(f.x, f.y), pkbf(f.z, f.w));
    };
    auto stageB = [&](u32 (*dst)[68], int k0) {
#pragma unroll
        for (int j = 0; j < 2; ++j) {
            int t  = tid + j * 128;
            int rp = t >> 4;
            int c4 = (t & 15) * 4;
            int r0 = k0 + 2 * rp, r1 = r0 + 1;
            int r0c = r0 < SPADR - 1 ? r0 : SPADR - 1;
            int r1c = r1 < SPADR - 1 ? r1 : SPADR - 1;
            float mA = (r0 < SREAL) ? 1.0f : 0.0f;
            float mB = (r1 < SREAL) ? 1.0f : 0.0f;
            const float4 f0 = *(const float4*)(V + (long)r0c * HID + c4);
            const float4 f1 = *(const float4*)(V + (long)r1c * HID + c4);
            uint4 u;
            u.x = pkbf(f0.x * mA, f1.x * mB);
            u.y = pkbf(f0.y * mA, f1.y * mB);
            u.z = pkbf(f0.z * mA, f1.z * mB);
            u.w = pkbf(f0.w * mA, f1.w * mB);
            *(uint4*)&dst[rp][c4] = u;
        }
    };

    stageA(As4[0], 0);
    stageB(Bs2[0], 0);
    __syncthreads();
    v8f acc = {};
    const int lr = lane & 15;
    const int p0 = (lane & 16) ? 4 : 0;
    int cur = 0;
    for (int kt = 0; kt < 3; ++kt) {
        if (kt + 1 < 3) { stageA(As4[cur ^ 1], (kt + 1) << 5); stageB(Bs2[cur ^ 1], (kt + 1) << 5); }
        v16bf fa = load_fa(As4[cur], lr, p0);
        v16bf fb = load_fb<68>(Bs2[cur], w * 16 + lr, p0);
        acc = wmma_bf16(fa, fb, acc);
        __syncthreads();
        cur ^= 1;
    }
    int colg  = w * 16 + lr;
    int rbase = (lane & 16) ? 8 : 0;
#pragma unroll
    for (int r = 0; r < 8; ++r) {
        int rowg = m0 + rbase + r;
        C[(long)rowg * HID + colg] = acc[r];
    }
}

// ---------------- layernorms ----------------
__global__ void __launch_bounds__(256)
ln_embed_kernel(const float* __restrict__ hpre, const float* __restrict__ g,
                const float* __restrict__ bb, const float* __restrict__ pos,
                float* __restrict__ hout) {
    const int r = blockIdx.x, b = blockIdx.y, tid = threadIdx.x;
    const float* in = hpre + ((long)b * RR + r) * HID;
    float* out = hout + ((long)b * RR + r) * HID;
    __shared__ float red[256];
    float x0 = in[tid], x1 = in[tid + 256], x2 = in[tid + 512];
    red[tid] = x0 + x1 + x2;
    __syncthreads();
    for (int off = 128; off > 0; off >>= 1) {
        if (tid < off) red[tid] += red[tid + off];
        __syncthreads();
    }
    float mean = red[0] * (1.0f / HID);
    __syncthreads();
    float d0 = x0 - mean, d1 = x1 - mean, d2 = x2 - mean;
    red[tid] = d0 * d0 + d1 * d1 + d2 * d2;
    __syncthreads();
    for (int off = 128; off > 0; off >>= 1) {
        if (tid < off) red[tid] += red[tid + off];
        __syncthreads();
    }
    float inv = rsqrtf(red[0] * (1.0f / HID) + 1e-12f);
    const float* pr = pos + (long)r * HID;
    out[tid]       = d0 * inv * g[tid]       + bb[tid]       + pr[tid];
    out[tid + 256] = d1 * inv * g[tid + 256] + bb[tid + 256] + pr[tid + 256];
    out[tid + 512] = d2 * inv * g[tid + 512] + bb[tid + 512] + pr[tid + 512];
}

__global__ void __launch_bounds__(256)
ln_mod_kernel(float* __restrict__ Ybase, long yBS,
              const float* __restrict__ gbase, const float* __restrict__ bbase,
              const int* __restrict__ args, int step, int lsel) {
    const int row = blockIdx.x, b = blockIdx.y, tid = threadIdx.x;
    float* Y = Ybase + (long)b * yBS + (long)row * HID;
    int prog = args[(b * NSTEP + step) * 2];
    prog = prog < 0 ? 0 : (prog >= NPROG ? NPROG - 1 : prog);
    long sel = (long)c_func_map[prog] * NLAY + lsel;
    const float* g  = gbase + sel * HID;
    const float* bb = bbase + sel * HID;
    __shared__ float red[256];
    float x0 = Y[tid], x1 = Y[tid + 256], x2 = Y[tid + 512];
    red[tid] = x0 + x1 + x2;
    __syncthreads();
    for (int off = 128; off > 0; off >>= 1) {
        if (tid < off) red[tid] += red[tid + off];
        __syncthreads();
    }
    float mean = red[0] * (1.0f / HID);
    __syncthreads();
    float d0 = x0 - mean, d1 = x1 - mean, d2 = x2 - mean;
    red[tid] = d0 * d0 + d1 * d1 + d2 * d2;
    __syncthreads();
    for (int off = 128; off > 0; off >>= 1) {
        if (tid < off) red[tid] += red[tid + off];
        __syncthreads();
    }
    float inv = rsqrtf(red[0] * (1.0f / HID) + 1e-12f);
    Y[tid]       = d0 * inv * g[tid]       + bb[tid];
    Y[tid + 256] = d1 * inv * g[tid + 256] + bb[tid + 256];
    Y[tid + 512] = d2 * inv * g[tid + 512] + bb[tid + 512];
}

// ---------------- sequence assembly ----------------
__global__ void __launch_bounds__(256)
build_x_kernel(const float* __restrict__ h, const float* __restrict__ arg_emb,
               const int* __restrict__ args, int step, float* __restrict__ x) {
    const int s = blockIdx.x, b = blockIdx.y, tid = threadIdx.x;
    float* xr = x + ((long)b * SPADR + s) * HID;
    if (s < RR) {
        const float* hr = h + ((long)b * RR + s) * HID;
        xr[tid] = hr[tid]; xr[tid + 256] = hr[tid + 256]; xr[tid + 512] = hr[tid + 512];
    } else if (s < SREAL) {
        int a = args[(b * NSTEP + step) * 2 + (s - RR)];
        a = a < 0 ? 0 : (a >= NARG ? NARG - 1 : a);
        const float* er = arg_emb + (long)a * HID;
        xr[tid] = er[tid]; xr[tid + 256] = er[tid + 256]; xr[tid + 512] = er[tid + 512];
    } else {
        xr[tid] = 0.0f; xr[tid + 256] = 0.0f; xr[tid + 512] = 0.0f;
    }
}

__global__ void __launch_bounds__(256)
update_h_kernel(const float* __restrict__ x, const int* __restrict__ args,
                int step, float* __restrict__ h) {
    const int r = blockIdx.x, b = blockIdx.y, tid = threadIdx.x;
    if (args[(b * NSTEP + step) * 2] < NPROG) {
        const float* xr = x + ((long)b * SPADR + r) * HID;
        float* hr = h + ((long)b * RR + r) * HID;
        hr[tid] = xr[tid]; hr[tid + 256] = xr[tid + 256]; hr[tid + 512] = xr[tid + 512];
    }
}

// ---------------- weight-normed head ----------------
__global__ void zero2_kernel(float* p) { if (threadIdx.x < 2) p[threadIdx.x] = 0.0f; }

__global__ void __launch_bounds__(256)
normsq_kernel(const float* __restrict__ v, long n, float* __restrict__ out) {
    __shared__ float red[256];
    float s = 0.0f;
    for (long i = (long)blockIdx.x * 256 + threadIdx.x; i < n; i += (long)gridDim.x * 256) {
        float t = v[i]; s += t * t;
    }
    red[threadIdx.x] = s;
    __syncthreads();
    for (int off = 128; off > 0; off >>= 1) {
        if (threadIdx.x < off) red[threadIdx.x] += red[threadIdx.x + off];
        __syncthreads();
    }
    if (threadIdx.x == 0) atomicAdd(out, red[0]);
}

__global__ void __launch_bounds__(256)
head1_kernel(const float* __restrict__ h, const float* __restrict__ V1,
             const float* __restrict__ g1, const float* __restrict__ b1,
             const float* __restrict__ norms, float* __restrict__ t) {
    int idx = blockIdx.x * 256 + threadIdx.x;
    if (idx >= BSZ * 2 * HID) return;
    int b = idx / (2 * HID), j = idx % (2 * HID);
    const float* h0 = h + (long)b * RR * HID;
    float s = 0.0f;
    for (int c = 0; c < HID; ++c) s += h0[c] * V1[(long)c * (2 * HID) + j];
    float val = s * g1[0] * rsqrtf(norms[0]) + b1[j];
    t[idx] = val > 0.0f ? val : 0.0f;
}

__global__ void __launch_bounds__(256)
head2_kernel(const float* __restrict__ t, const float* __restrict__ V2,
             const float* __restrict__ g2, const float* __restrict__ b2,
             const float* __restrict__ norms, float* __restrict__ pred) {
    int idx = blockIdx.x * 256 + threadIdx.x;
    if (idx >= BSZ * NLABEL) return;
    int b = idx / NLABEL, j = idx % NLABEL;
    float s = 0.0f;
    for (int c = 0; c < 2 * HID; ++c) s += t[(long)b * 2 * HID + c] * V2[(long)c * NLABEL + j];
    pred[idx] = s * g2[0] * rsqrtf(norms[1]) + b2[j];
}

// ---------------- workspace layout ----------------
#define WS_X    0L
#define WS_Q    491520L
#define WS_K    983040L
#define WS_V    1474560L
#define WS_SC   1966080L
#define WS_CTX  2703360L
#define WS_A    3194880L
#define WS_I    3686400L
#define WS_HP   5652480L
#define WS_T    6045696L
#define WS_N    6057984L
#define WS_PK   6057988L          // start of packed-weight region (16B aligned)
#define PK_Q    0L
#define PK_K    7077888L
#define PK_V    14155776L
#define PK_O    21233664L
#define PK_I    28311552L
#define PK_O2   56623104L
#define PK_IMG  84934656L
#define PK_END  85721088L
#define WS_NEED_BYTES ((WS_PK + PK_END) * 4L)   // ~367 MB total

extern "C" void kernel_launch(void* const* d_in, const int* in_sizes, int n_in,
                              void* d_out, int out_size, void* d_ws, size_t ws_size,
                              hipStream_t stream) {
    (void)in_sizes; (void)n_in; (void)out_size;
    const float* features = (const float*)d_in[0];
    const float* spatials = (const float*)d_in[1];
    const int*   args     = (const int*)d_in[3];
    const float* W_img    = (const float*)d_in[4];
    const float* b_img    = (const float*)d_in[5];
    const float* W_loc    = (const float*)d_in[6];
    const float* b_loc    = (const float*)d_in[7];
    const float* ln_img_g = (const float*)d_in[8];
    const float* ln_img_b = (const float*)d_in[9];
    const float* arg_emb  = (const float*)d_in[10];
    const float* pos_emb  = (const float*)d_in[11];
    const float* Wq  = (const float*)d_in[12];
    const float* Wk  = (const float*)d_in[13];
    const float* Wv  = (const float*)d_in[14];
    const float* Wo  = (const float*)d_in[15];
    const float* bq  = (const float*)d_in[16];
    const float* bk  = (const float*)d_in[17];
    const float* bv  = (const float*)d_in[18];
    const float* bo  = (const float*)d_in[19];
    const float* ln1_g = (const float*)d_in[20];
    const float* ln1_b = (const float*)d_in[21];
    const float* Wi  = (const float*)d_in[22];
    const float* bi  = (const float*)d_in[23];
    const float* Wo2 = (const float*)d_in[24];
    const float* bo2 = (const float*)d_in[25];
    const float* ln2_g = (const float*)d_in[26];
    const float* ln2_b = (const float*)d_in[27];
    const float* V1 = (const float*)d_in[28];
    const float* g1 = (const float*)d_in[29];
    const float* b1 = (const float*)d_in[30];
    const float* V2 = (const float*)d_in[31];
    const float* g2 = (const float*)d_in[32];
    const float* b2 = (const float*)d_in[33];

    float* out  = (float*)d_out;
    float* h    = out;
    float* pred = out + (long)BSZ * RR * HID;

    float* ws    = (float*)d_ws;
    float* x     = ws + WS_X;
    float* q     = ws + WS_Q;
    float* k     = ws + WS_K;
    float* v     = ws + WS_V;
    float* sc    = ws + WS_SC;
    float* ctx   = ws + WS_CTX;
    float* a     = ws + WS_A;
    float* inter = ws + WS_I;
    float* hpre  = ws + WS_HP;
    float* t     = ws + WS_T;
    float* norms = ws + WS_N;

    const bool packed = ws_size >= (size_t)WS_NEED_BYTES;
    u32* pk   = (u32*)(ws + WS_PK);
    u32* pWq  = packed ? pk + PK_Q   : nullptr;
    u32* pWk  = packed ? pk + PK_K   : nullptr;
    u32* pWv  = packed ? pk + PK_V   : nullptr;
    u32* pWo  = packed ? pk + PK_O   : nullptr;
    u32* pWi  = packed ? pk + PK_I   : nullptr;
    u32* pWo2 = packed ? pk + PK_O2  : nullptr;
    u32* pWim = packed ? pk + PK_IMG : nullptr;

    if (packed) {
        const long rQ  = (long)NMOD * NLAY * HID;
        const long rO2 = (long)NMOD * NLAY * FFD;
        pack_weights_kernel<<<2048, 256, 0, stream>>>(Wq,  pWq,  rQ,  HID);
        pack_weights_kernel<<<2048, 256, 0, stream>>>(Wk,  pWk,  rQ,  HID);
        pack_weights_kernel<<<2048, 256, 0, stream>>>(Wv,  pWv,  rQ,  HID);
        pack_weights_kernel<<<2048, 256, 0, stream>>>(Wo,  pWo,  rQ,  HID);
        pack_weights_kernel<<<4096, 256, 0, stream>>>(Wi,  pWi,  rQ,  FFD);
        pack_weights_kernel<<<4096, 256, 0, stream>>>(Wo2, pWo2, rO2, HID);
        pack_weights_kernel<<<512,  256, 0, stream>>>(W_img, pWim, (long)VFD, HID);
    }

    const long SH = (long)SPADR * HID;
    const long SF = (long)SPADR * FFD;
    const dim3 gModH(HID / 128, 3, BSZ);
    const dim3 gModF(FFD / 128, 3, BSZ);

    wmma_gemm_kernel<<<dim3(HID / 128, (BSZ * RR) / 32, 1), 256, 0, stream>>>(
        features, 0L, VFD, BSZ * RR, BSZ * RR,
        W_img, pWim, 0L, HID, VFD,
        b_img, 0L, b_loc,
        nullptr, 0L,
        spatials, W_loc,
        hpre, 0L, HID,
        nullptr, 0, 0, 0);
    ln_embed_kernel<<<dim3(RR, BSZ), 256, 0, stream>>>(hpre, ln_img_g, ln_img_b, pos_emb, h);

    const float scale = 0.125f;
    for (int step = 0; step < NSTEP; ++step) {
        build_x_kernel<<<dim3(SPADR, BSZ), 256, 0, stream>>>(h, arg_emb, args, step, x);
        for (int l = 0; l < NLAY; ++l) {
            wmma_gemm_kernel<<<gModH, 256, 0, stream>>>(
                x, SH, HID, SREAL, SPADR, Wq, pWq, (long)HID * HID, HID, HID,
                bq, (long)HID, nullptr, nullptr, 0L, nullptr, nullptr,
                q, SH, HID, args, step, l, 0);
            wmma_gemm_kernel<<<gModH, 256, 0, stream>>>(
                x, SH, HID, SREAL, SPADR, Wk, pWk, (long)HID * HID, HID, HID,
                bk, (long)HID, nullptr, nullptr, 0L, nullptr, nullptr,
                k, SH, HID, args, step, l, 0);
            wmma_gemm_kernel<<<gModH, 256, 0, stream>>>(
                x, SH, HID, SREAL, SPADR, Wv, pWv, (long)HID * HID, HID, HID,
                bv, (long)HID, nullptr, nullptr, 0L, nullptr, nullptr,
                v, SH, HID, args, step, l, 0);
            attn_scores_kernel<<<dim3(2, SPADR / 16, BSZ * NHEAD), 128, 0, stream>>>(q, k, sc, scale);
            softmax_kernel<<<dim3(SREAL, BSZ * NHEAD), 128, 0, stream>>>(sc);
            attn_ctx_kernel<<<dim3(1, SPADR / 16, BSZ * NHEAD), 128, 0, stream>>>(sc, v, ctx);
            wmma_gemm_kernel<<<gModH, 256, 0, stream>>>(
                ctx, SH, HID, SREAL, SPADR, Wo, pWo, (long)HID * HID, HID, HID,
                bo, (long)HID, nullptr, x, SH, nullptr, nullptr,
                a, SH, HID, args, step, l, 0);
            ln_mod_kernel<<<dim3(SREAL, BSZ), 256, 0, stream>>>(a, SH, ln1_g, ln1_b, args, step, l);
            wmma_gemm_kernel<<<gModF, 256, 0, stream>>>(
                a, SH, HID, SREAL, SPADR, Wi, pWi, (long)HID * FFD, FFD, HID,
                bi, (long)FFD, nullptr, nullptr, 0L, nullptr, nullptr,
                inter, SF, FFD, args, step, l, 1);
            wmma_gemm_kernel<<<gModH, 256, 0, stream>>>(
                inter, SF, FFD, SREAL, SPADR, Wo2, pWo2, (long)FFD * HID, HID, FFD,
                bo2, (long)HID, nullptr, a, SH, nullptr, nullptr,
                x, SH, HID, args, step, l, 0);
            ln_mod_kernel<<<dim3(SREAL, BSZ), 256, 0, stream>>>(x, SH, ln2_g, ln2_b, args, step, l);
        }
        update_h_kernel<<<dim3(RR, BSZ), 256, 0, stream>>>(x, args, step, h);
    }

    zero2_kernel<<<1, 32, 0, stream>>>(norms);
    normsq_kernel<<<dim3(256), 256, 0, stream>>>(V1, (long)HID * 2 * HID, norms + 0);
    normsq_kernel<<<dim3(64), 256, 0, stream>>>(V2, (long)2 * HID * NLABEL, norms + 1);
    head1_kernel<<<dim3((BSZ * 2 * HID + 255) / 256), 256, 0, stream>>>(h, V1, g1, b1, norms, t);
    head2_kernel<<<dim3(1), 256, 0, stream>>>(t, V2, g2, b2, norms, pred);
}
